// MPGKScoreOperator_83425444757510
// MI455X (gfx1250) — matrix operand order repaired and checked by reading, not compiled, not run on hardware
//
#include <hip/hip_runtime.h>
#include <hip/hip_bf16.h>
#include <cstdint>

typedef __attribute__((ext_vector_type(16))) _Float16 v16h;
typedef __attribute__((ext_vector_type(8)))  _Float16 h8;
typedef __attribute__((ext_vector_type(8)))  float    v8f;
typedef __attribute__((ext_vector_type(4)))  float    v4f;
typedef __attribute__((ext_vector_type(4)))  unsigned int u32x4;
typedef __attribute__((ext_vector_type(8)))  int      i32x8;
typedef __attribute__((ext_vector_type(4)))  int      i32x4;

#define CAT16(lo, hi) __builtin_shufflevector((lo), (hi), 0,1,2,3,4,5,6,7,8,9,10,11,12,13,14,15)

__device__ __forceinline__ float silu_f(float x) { return x / (1.0f + __expf(-x)); }
// tanh-approx gelu: x * sigmoid(2*sqrt(2/pi)*(x + 0.044715 x^3))
__device__ __forceinline__ float gelu_f(float x) {
  float u = 1.5957691216057308f * (x + 0.044715f * x * x * x);
  return x / (1.0f + __expf(-u));
}

// ---------------------------------------------------------------------------
// GNO kernel integral transform:
//   out[b,i,c] = act( add[b,i,c] + (1/cnt_i) * sum_j mask_ij * k_ij[c] * f[b,j,c] )
//   k_ij = gelu(aY[j,:] + bX[i,:]) @ W2 + b2      (H=64, C=128, B=4)
// Block: 256 threads (8 waves). Block = one 16-point x tile; wave w = c-tile w.
// aY j-tile is staged into LDS by the Tensor Data Mover (tensor_load_to_lds).
// ---------------------------------------------------------------------------
__global__ __launch_bounds__(256)
void gno_kernel(const float* __restrict__ xyX, int Nx,
                const float* __restrict__ xyY, int Ny,
                const float* __restrict__ aY,    // (Ny,64)
                const float* __restrict__ bX,    // (Nx,64)
                const float* __restrict__ W2,    // (64,128)
                const float* __restrict__ b2,    // (128)
                const float* __restrict__ fy,    // (4,Ny,128)
                const float* __restrict__ addB,  // nullable (4,Nx,128)
                float* __restrict__ outB,        // (4,Nx,128)
                float r2, int actSilu)
{
  __shared__ _Float16 W2t[128 * 64];   // transposed [c][h], f16 (16 KB)
  __shared__ _Float16 Hlds[256 * 64];  // [pair p = ii*16+jj][h], f16 (32 KB)
  __shared__ float aYl[16 * 64];
  __shared__ float bXl[16 * 64];
  __shared__ float yl[32];
  __shared__ unsigned int maskLds[16];
  __shared__ int cntLds[16];

  const int t = threadIdx.x;
  const int lane = t & 31;
  const int wv = t >> 5;                  // wave id == c-tile
  const int i_base = blockIdx.x * 16;

  // Stage W2 (64x128 f32, [h][c]) -> W2t[c][h] f16
#pragma unroll
  for (int k = 0; k < 32; ++k) {
    int idx = t + k * 256;
    int h = idx >> 7, c = idx & 127;
    W2t[c * 64 + h] = (_Float16)W2[idx];
  }
  ((v4f*)bXl)[t] = ((const v4f*)(bX + (size_t)i_base * 64))[t];
  if (t < 16) cntLds[t] = 0;
  __syncthreads();

  const int lo16 = lane & 15;
  const int hi8 = (lane < 16) ? 0 : 8;
  const int cl = wv * 16 + lo16;          // this lane's channel
  const float b2l = b2[cl];

  // B fragments (32x16 f16): lanes 0-15 hold K 0..15, lanes 16-31 hold K 16..31
  const h8* wrow = (const h8*)&W2t[cl * 64];
  const int kc = (lane < 16) ? 0 : 2;
  const v16h Bf0 = CAT16(wrow[kc], wrow[kc + 1]);          // K chunk 0..31
  const v16h Bf1 = CAT16(wrow[kc + 4], wrow[kc + 5]);      // K chunk 32..63

  const int iiT = wv * 2 + (lane >> 4);   // this thread's ii (== t>>4)
  const float xix = xyX[(i_base + iiT) * 2 + 0];
  const float xiy = xyX[(i_base + iiT) * 2 + 1];

  // --- TDM descriptor pieces for the aY j-tile (1024 contiguous f32 = 4KB) ---
  const unsigned long long aYbase = (unsigned long long)(uintptr_t)aY;
  const unsigned int ldsA = (unsigned int)(uintptr_t)aYl;
  i32x8 g1;
  g1[0] = 0x00020000;            // workgroup_mask=0, data_size=2 (4 bytes)
  g1[1] = (int)(1024u << 16);    // tensor_dim0 = 1024 (bits 79:48, low 16 here)
  g1[2] = (int)(1u << 16);       // tensor_dim1 = 1
  g1[3] = (int)(1024u << 16);    // tile_dim0 = 1024
  g1[4] = 1;                     // tile_dim1 = 1
  g1[5] = 1024;                  // tensor_dim0_stride = 1024
  g1[6] = 0;
  g1[7] = 0;
  const i32x4 gz4 = {0, 0, 0, 0};
  const i32x8 gz8 = {0, 0, 0, 0, 0, 0, 0, 0};

  float acc[4][16];
#pragma unroll
  for (int b = 0; b < 4; ++b)
#pragma unroll
    for (int m = 0; m < 16; ++m) acc[b][m] = 0.0f;

  for (int j_base = 0; j_base < Ny; j_base += 16) {
    __syncthreads();
    if (wv == 0) {
      yl[lane] = xyY[j_base * 2 + lane];
      // async DMA: aY[j_base*64 .. +1024) -> aYl  (issued once, wave-level)
      unsigned long long ga = aYbase + (unsigned long long)j_base * 256ull;
      u32x4 g0;
      g0[0] = 1u;                                        // count=1, user mode
      g0[1] = ldsA;                                      // lds_addr
      g0[2] = (unsigned int)ga;                          // global_addr lo
      g0[3] = (unsigned int)(ga >> 32) | (2u << 30);     // global_addr hi | type=2
      __builtin_amdgcn_tensor_load_to_lds(g0, g1, gz4, gz4, gz8, 0);
      __builtin_amdgcn_s_wait_tensorcnt(0);
    }
    __syncthreads();

    // radius mask + neighbor counts (wave covers pairs for ii = 2w, 2w+1)
    float dx = xix - yl[lo16 * 2 + 0];
    float dy = xiy - yl[lo16 * 2 + 1];
    const bool selfPass = (dx * dx + dy * dy <= r2);
    {
      unsigned int bal = (unsigned int)__ballot(selfPass);
      if (lane == 0) {
        maskLds[wv * 2 + 0] = bal & 0xffffu;
        maskLds[wv * 2 + 1] = bal >> 16;
        cntLds[wv * 2 + 0] += __popc(bal & 0xffffu);
        cntLds[wv * 2 + 1] += __popc(bal >> 16);
      }
    }

    // build H = mask * gelu(aY + bX) f16, one pair-row per thread
    // (row pre-masked so the WMMA output needs no per-element mask select)
    {
      const int jj = t & 15, ii = t >> 4;
      const unsigned int pm = selfPass ? 0xffffffffu : 0u;
      const float* arow = &aYl[jj * 64];
      const float* brow = &bXl[ii * 64];
      h8* hrow = (h8*)&Hlds[t * 64];
#pragma unroll
      for (int hc = 0; hc < 8; ++hc) {
        h8 o;
#pragma unroll
        for (int e = 0; e < 8; ++e)
          o[e] = (_Float16)gelu_f(arow[hc * 8 + e] + brow[hc * 8 + e]);
        i32x4 oi = __builtin_bit_cast(i32x4, o);
        oi &= (int)pm;
        hrow[hc] = __builtin_bit_cast(h8, oi);
      }
    }
    __syncthreads();

    // f values for this c-lane: f[b][j_base + v + hi8][cl]  (L2-resident)
    float fr[8][4];
#pragma unroll
    for (int v = 0; v < 8; ++v) {
      const int j = j_base + v + hi8;
#pragma unroll
      for (int b = 0; b < 4; ++b)
        fr[v][b] = fy[((size_t)b * Ny + j) * 128 + cl];
    }
    if (j_base + 16 < Ny)  // prefetch next tile's f region (global_prefetch_b8)
      __builtin_prefetch(&fy[((size_t)(j_base + 16 + hi8)) * 128 + cl], 0, 1);

    const int c0 = hi8 >> 2;  // A-frag chunk base (contiguous K order, matches B)
#pragma unroll
    for (int m = 0; m < 16; ++m) {
      const h8* hrow = (const h8*)&Hlds[(m * 16 + lo16) * 64];
      v16h A0 = CAT16(hrow[c0], hrow[c0 + 1]);       // K 0..31
      v16h A1 = CAT16(hrow[c0 + 4], hrow[c0 + 5]);   // K 32..63
      v8f kacc = {};
      kacc = __builtin_amdgcn_wmma_f32_16x16x32_f16(false, A0, false, Bf0,
                                                    (short)0, kacc, false, false);
      kacc = __builtin_amdgcn_wmma_f32_16x16x32_f16(false, A1, false, Bf1,
                                                    (short)0, kacc, false, false);
      const unsigned int m16 = maskLds[m];
#pragma unroll
      for (int v = 0; v < 8; ++v) {
        // H rows are pre-masked; only the b2 term still needs the mask bit
        float mf = (float)((m16 >> (v + hi8)) & 1u);
        float kk = fmaf(mf, b2l, kacc[v]);
#pragma unroll
        for (int b = 0; b < 4; ++b)
          acc[b][m] = fmaf(kk, fr[v][b], acc[b][m]);
      }
    }
  }
  __syncthreads();

#pragma unroll
  for (int m = 0; m < 16; ++m) {
    int cnt = cntLds[m]; if (cnt < 1) cnt = 1;
    const float inv = 1.0f / (float)cnt;
#pragma unroll
    for (int b = 0; b < 4; ++b) {
      float s = acc[b][m] + __shfl_xor(acc[b][m], 16, 32);  // merge jj halves
      if (lane < 16) {
        size_t off = ((size_t)b * Nx + i_base + m) * 128 + cl;
        float o = s * inv;
        if (addB) o += addB[off];
        if (actSilu) o = silu_f(o);
        outB[off] = o;
      }
    }
  }
}

// ---------------------------------------------------------------------------
// small helper kernels
// ---------------------------------------------------------------------------
__global__ void zero_kernel(float* p, int n) {
  int i = blockIdx.x * blockDim.x + threadIdx.x;
  if (i < n) p[i] = 0.0f;
}

__global__ void pack_coords_kernel(const float* __restrict__ coords,
                                   float* xy0, float* xy1, float* xy2) {
  int i = blockIdx.x * blockDim.x + threadIdx.x;
  if (i < 1024) { xy0[2 * i] = coords[i];       xy0[2 * i + 1] = coords[1024 + i]; }
  if (i < 512)  { xy1[2 * i] = coords[2 * i];   xy1[2 * i + 1] = coords[1024 + 2 * i]; }
  if (i < 256)  { xy2[2 * i] = coords[4 * i];   xy2[2 * i + 1] = coords[1024 + 4 * i]; }
}

// aY[j,h] = y0*W1[0,h] + y1*W1[1,h]         (xpart=0)
// bX[i,h] = x0*W1[2,h] + x1*W1[3,h] + b1[h] (xpart=1)
__global__ void ab_kernel(const float* __restrict__ xy, int N,
                          const float* __restrict__ W1, const float* __restrict__ b1,
                          int xpart, float* __restrict__ out) {
  int idx = blockIdx.x * blockDim.x + threadIdx.x;
  if (idx >= N * 64) return;
  int n = idx >> 6, h = idx & 63;
  int c0 = xpart ? 2 : 0;
  float v = xy[2 * n] * W1[c0 * 64 + h] + xy[2 * n + 1] * W1[(c0 + 1) * 64 + h];
  if (xpart) v += b1[h];
  out[idx] = v;
}

__global__ void lift_kernel(const float* __restrict__ samples,
                            const float* __restrict__ w, const float* __restrict__ b,
                            float* __restrict__ out) {
  int idx = blockIdx.x * blockDim.x + threadIdx.x;  // 4*1024*128
  if (idx >= 4 * 1024 * 128) return;
  int c = idx & 127, n = (idx >> 7) & 1023, bb = idx >> 17;
  out[idx] = samples[bb * 1024 + n] * w[c] + b[c];
}

// FiLM shift/scale (sigma-dependent only; hoisted out of the iteration loop)
__global__ __launch_bounds__(256)
void film_pre_kernel(const float* __restrict__ sigma,
                     const float* __restrict__ e1w, const float* __restrict__ e1b,
                     const float* __restrict__ e2w, const float* __restrict__ e2b,
                     float* __restrict__ shift, float* __restrict__ scale) {
  __shared__ float emb[4 * 128];
  __shared__ float hbuf[4 * 256];
  int t = threadIdx.x;
  {
    int b = t >> 6, k = t & 63;
    float fr = __powf(1.0f / 10000.0f, (float)k / 63.0f);
    float xf = sigma[b] * fr;
    emb[b * 128 + k] = __sinf(xf);
    emb[b * 128 + 64 + k] = __cosf(xf);
  }
  __syncthreads();
  for (int b = 0; b < 4; ++b) {
    float s = e1b[t];
    for (int k = 0; k < 128; ++k) s += emb[b * 128 + k] * e1w[k * 256 + t];
    hbuf[b * 256 + t] = silu_f(s);
  }
  __syncthreads();
  for (int b = 0; b < 4; ++b) {
    float s = e2b[t];
    for (int k = 0; k < 256; ++k) s += hbuf[b * 256 + k] * e2w[k * 256 + t];
    s = silu_f(s);
    if (t < 128) shift[b * 128 + t] = s;
    else         scale[b * 128 + (t - 128)] = s;
  }
}

// g = silu(shift + (groupnorm(v)*gn_scale+gn_bias)*(scale+1)); one row per block
__global__ __launch_bounds__(128)
void film_apply_kernel(const float* __restrict__ v,
                       const float* __restrict__ shift, const float* __restrict__ scale,
                       const float* __restrict__ gns, const float* __restrict__ gnb,
                       int N, float* __restrict__ out) {
  __shared__ float row[128];
  int c = threadIdx.x;
  int r = blockIdx.x;          // r = b*N + n
  int b = r / N;
  size_t off = (size_t)r * 128 + c;
  float x = v[off];
  row[c] = x;
  __syncthreads();
  int g = c & ~3;              // groups of C/G = 4 channels
  float a0 = row[g], a1 = row[g + 1], a2 = row[g + 2], a3 = row[g + 3];
  float m = 0.25f * (a0 + a1 + a2 + a3);
  float var = 0.25f * ((a0 - m) * (a0 - m) + (a1 - m) * (a1 - m) +
                       (a2 - m) * (a2 - m) + (a3 - m) * (a3 - m));
  float nrm = (x - m) * rsqrtf(var + 1e-5f);
  float y = nrm * gns[c] + gnb[c];
  float o = shift[b * 128 + c] + y * (scale[b * 128 + c] + 1.0f);
  out[off] = silu_f(o);
}

// out = v @ W + bias (+ add)
__global__ void linw_kernel(const float* __restrict__ v, const float* __restrict__ W,
                            const float* __restrict__ bias, const float* __restrict__ add,
                            int rows, float* __restrict__ out) {
  int idx = blockIdx.x * blockDim.x + threadIdx.x;
  if (idx >= rows * 128) return;
  int c = idx & 127, r = idx >> 7;
  const float* vr = v + (size_t)r * 128;
  float s = bias[c];
  for (int k = 0; k < 128; ++k) s += vr[k] * W[k * 128 + c];
  if (add) s += add[idx];
  out[idx] = s;
}

__global__ void proj_kernel(const float* __restrict__ v, const float* __restrict__ w,
                            const float* __restrict__ b, float* __restrict__ out, int total) {
  int idx = blockIdx.x * blockDim.x + threadIdx.x;
  if (idx >= total) return;
  const float* vr = v + (size_t)idx * 128;
  float s = b[0];
  for (int k = 0; k < 128; ++k) s += vr[k] * w[k];
  out[idx] = s;
}

// ---------------------------------------------------------------------------
// host orchestration
// ---------------------------------------------------------------------------
enum {
  IN_COORDS = 0, IN_SAMPLES = 1, IN_SIGMA = 2,
  L0_Wb = 3, L0_Ww, L0_E1b, L0_E1w, L0_E2b, L0_E2w,
  L0_DN_1b, L0_DN_1w, L0_DN_2b, L0_DN_2w,
  L0_LL_1b, L0_LL_1w, L0_LL_2b, L0_LL_2w,
  L0_UP_1b, L0_UP_1w, L0_UP_2b, L0_UP_2w,
  L0_GNb, L0_GNs,
  L1_Wb, L1_Ww, L1_E1b, L1_E1w, L1_E2b, L1_E2w,
  L1_DN_1b, L1_DN_1w, L1_DN_2b, L1_DN_2w,
  L1_LL_1b, L1_LL_1w, L1_LL_2b, L1_LL_2w,
  L1_UP_1b, L1_UP_1w, L1_UP_2b, L1_UP_2w,
  L1_GNb, L1_GNs,
  L2_Wb, L2_Ww, L2_E1b, L2_E1w, L2_E2b, L2_E2w,
  L2_LL_1b, L2_LL_1w, L2_LL_2b, L2_LL_2w,
  L2_GNb, L2_GNs,
  LIFT_b, LIFT_w, PROJ_b, PROJ_w
};

extern "C" void kernel_launch(void* const* d_in, const int* in_sizes, int n_in,
                              void* d_out, int out_size, void* d_ws, size_t ws_size,
                              hipStream_t stream) {
  (void)in_sizes; (void)n_in; (void)out_size; (void)ws_size;
  const float* coords  = (const float*)d_in[IN_COORDS];
  const float* samples = (const float*)d_in[IN_SAMPLES];
  const float* sigma   = (const float*)d_in[IN_SIGMA];
  auto P = [&](int i) { return (const float*)d_in[i]; };

  float* w = (float*)d_ws;
  size_t off = 0;
  auto alloc = [&](size_t n) { float* p = w + off; off += n; return p; };

  float* xy0 = alloc(2048);
  float* xy1 = alloc(1024);
  float* xy2 = alloc(512);
  float* aYd1 = alloc(1024 * 64); float* bXd1 = alloc(512 * 64);
  float* aYd2 = alloc(512 * 64);  float* bXd2 = alloc(256 * 64);
  float* aYu1 = alloc(256 * 64);  float* bXu1 = alloc(512 * 64);
  float* aYu0 = alloc(512 * 64);  float* bXu0 = alloc(1024 * 64);
  float* aYl0 = alloc(1024 * 64); float* bXl0 = alloc(1024 * 64);
  float* aYl1 = alloc(512 * 64);  float* bXl1 = alloc(512 * 64);
  float* aYl2 = alloc(256 * 64);  float* bXl2 = alloc(256 * 64);
  float* sh0 = alloc(512); float* sc0 = alloc(512);
  float* sh1 = alloc(512); float* sc1 = alloc(512);
  float* sh2 = alloc(512); float* sc2 = alloc(512);
  float* vd0 = alloc(4 * 1024 * 128);
  float* vd1 = alloc(4 * 512 * 128);
  float* vd2 = alloc(4 * 256 * 128);
  float* vu0 = alloc(4 * 1024 * 128);
  float* vu1 = alloc(4 * 512 * 128);
  float* vu2 = alloc(4 * 256 * 128);
  float* gbuf = alloc(4 * 1024 * 128);
  float* ladd = alloc(4 * 1024 * 128);
  float* tmpU = alloc(4 * 1024 * 128);

  const float r035 = 0.035f, r07 = 0.07f, r14 = 0.14f;

  pack_coords_kernel<<<4, 256, 0, stream>>>(coords, xy0, xy1, xy2);

  auto ab = [&](const float* xy, int N, int wi, int bi, int xpart, float* out_) {
    ab_kernel<<<(N * 64 + 255) / 256, 256, 0, stream>>>(xy, N, P(wi),
                                                        bi >= 0 ? P(bi) : nullptr, xpart, out_);
  };
  ab(xy0, 1024, L0_DN_1w, -1, 0, aYd1);      ab(xy1, 512,  L0_DN_1w, L0_DN_1b, 1, bXd1);
  ab(xy1, 512,  L1_DN_1w, -1, 0, aYd2);      ab(xy2, 256,  L1_DN_1w, L1_DN_1b, 1, bXd2);
  ab(xy2, 256,  L1_UP_1w, -1, 0, aYu1);      ab(xy1, 512,  L1_UP_1w, L1_UP_1b, 1, bXu1);
  ab(xy1, 512,  L0_UP_1w, -1, 0, aYu0);      ab(xy0, 1024, L0_UP_1w, L0_UP_1b, 1, bXu0);
  ab(xy0, 1024, L0_LL_1w, -1, 0, aYl0);      ab(xy0, 1024, L0_LL_1w, L0_LL_1b, 1, bXl0);
  ab(xy1, 512,  L1_LL_1w, -1, 0, aYl1);      ab(xy1, 512,  L1_LL_1w, L1_LL_1b, 1, bXl1);
  ab(xy2, 256,  L2_LL_1w, -1, 0, aYl2);      ab(xy2, 256,  L2_LL_1w, L2_LL_1b, 1, bXl2);

  lift_kernel<<<(4 * 1024 * 128) / 256, 256, 0, stream>>>(samples, P(LIFT_w), P(LIFT_b), vd0);

  film_pre_kernel<<<1, 256, 0, stream>>>(sigma, P(L0_E1w), P(L0_E1b), P(L0_E2w), P(L0_E2b), sh0, sc0);
  film_pre_kernel<<<1, 256, 0, stream>>>(sigma, P(L1_E1w), P(L1_E1b), P(L1_E2w), P(L1_E2b), sh1, sc1);
  film_pre_kernel<<<1, 256, 0, stream>>>(sigma, P(L2_E1w), P(L2_E1b), P(L2_E2w), P(L2_E2b), sh2, sc2);

  zero_kernel<<<(4 * 512 * 128) / 256, 256, 0, stream>>>(vu1, 4 * 512 * 128);
  zero_kernel<<<(4 * 256 * 128) / 256, 256, 0, stream>>>(vu2, 4 * 256 * 128);

  auto gno = [&](const float* xyX, int Nx, const float* xyY, int Ny,
                 const float* aY, const float* bX, int w2i, int b2i,
                 const float* fy, const float* addB, float* outB, float r, int act) {
    gno_kernel<<<Nx / 16, 256, 0, stream>>>(xyX, Nx, xyY, Ny, aY, bX, P(w2i), P(b2i),
                                            fy, addB, outB, r * r, act);
  };

  for (int it = 0; it < 3; ++it) {
    // ---- down pass ----
    gno(xy1, 512, xy0, 1024, aYd1, bXd1, L0_DN_2w, L0_DN_2b, vd0, vu1, vd1, r07, 1);
    gno(xy2, 256, xy1, 512,  aYd2, bXd2, L1_DN_2w, L1_DN_2b, vd1, vu2, vd2, r14, 1);

    // ---- bottom level (L=2) ----
    film_apply_kernel<<<4 * 256, 128, 0, stream>>>(vd2, sh2, sc2, P(L2_GNs), P(L2_GNb), 256, gbuf);
    linw_kernel<<<(4 * 256 * 128) / 256, 256, 0, stream>>>(vd2, P(L2_Ww), P(L2_Wb), nullptr, 4 * 256, ladd);
    gno(xy2, 256, xy2, 256, aYl2, bXl2, L2_LL_2w, L2_LL_2b, gbuf, ladd, vu2, r14, 1);

    // ---- up pass l=1 ----
    gno(xy1, 512, xy2, 256, aYu1, bXu1, L1_UP_2w, L1_UP_2b, vu2, nullptr, tmpU, r14, 0);
    film_apply_kernel<<<4 * 512, 128, 0, stream>>>(vd1, sh1, sc1, P(L1_GNs), P(L1_GNb), 512, gbuf);
    linw_kernel<<<(4 * 512 * 128) / 256, 256, 0, stream>>>(vd1, P(L1_Ww), P(L1_Wb), tmpU, 4 * 512, ladd);
    gno(xy1, 512, xy1, 512, aYl1, bXl1, L1_LL_2w, L1_LL_2b, gbuf, ladd, vu1, r07, 1);

    // ---- up pass l=0 ----
    gno(xy0, 1024, xy1, 512, aYu0, bXu0, L0_UP_2w, L0_UP_2b, vu1, nullptr, tmpU, r07, 0);
    film_apply_kernel<<<4 * 1024, 128, 0, stream>>>(vd0, sh0, sc0, P(L0_GNs), P(L0_GNb), 1024, gbuf);
    linw_kernel<<<(4 * 1024 * 128) / 256, 256, 0, stream>>>(vd0, P(L0_Ww), P(L0_Wb), tmpU, 4 * 1024, ladd);
    gno(xy0, 1024, xy0, 1024, aYl0, bXl0, L0_LL_2w, L0_LL_2b, gbuf, ladd, vu0, r035, 1);
  }

  proj_kernel<<<(4096 + 255) / 256, 256, 0, stream>>>(vu0, P(PROJ_w), P(PROJ_b), (float*)d_out, 4096);
}
